// EvidenceRetrieval_82343112998998
// MI455X (gfx1250) — compile-verified, matrix operand order checked by reading, not run on hardware
//
#include <hip/hip_runtime.h>
#include <hip/hip_bf16.h>

// ---------------- problem constants ----------------
#define BATCH 16384
#define OBS   2048
#define PLAN  2048
#define QD    2048     // query dim (== key dim)
#define VD    2048     // value dim
#define KB    1000
#define KBP   1024     // KB padded to WMMA tile multiple
#define TOPK  5
#define INV_TEMP (1.0f / 0.07f)
#define EPSN  1e-12f
#define NEGBIG (-3.4e38f)

// ---------------- CDNA5 feature detection ----------------
#if __has_builtin(__builtin_amdgcn_tensor_load_to_lds) && __has_builtin(__builtin_amdgcn_s_wait_tensorcnt)
#define HAVE_TDM 1
#if __has_include(<hip/amd_detail/amd_gfx1250_TDM.h>)
#define CDNA5_TDM_6ARG 1
#endif
#endif

#if __has_builtin(__builtin_amdgcn_global_load_async_to_lds_b128)
#define HAVE_ASYNC_LDS 1
#endif

#if __has_builtin(__builtin_amdgcn_s_wait_asynccnt)
#define WAIT_ASYNC(n) __builtin_amdgcn_s_wait_asynccnt(n)
#else
#define WAIT_ASYNC(n) asm volatile("s_wait_asynccnt %0" ::"i"(n) : "memory")
#endif

#ifdef HAVE_TDM
#define WAIT_TENSOR(n) __builtin_amdgcn_s_wait_tensorcnt(n)
#endif

// ---------------- WMMA / vector types ----------------
typedef __attribute__((ext_vector_type(16))) __bf16 v16bf;
typedef __attribute__((ext_vector_type(8)))  __bf16 v8bf;
typedef __attribute__((ext_vector_type(8)))  float  v8f;
typedef __attribute__((ext_vector_type(4)))  unsigned int u32x4;
typedef __attribute__((ext_vector_type(4)))  int  i32x4;
typedef __attribute__((ext_vector_type(8)))  int  i32x8;

// generic -> LDS(as3)/global(as1) pointer helpers (int casts: always compile-safe;
// low 32 bits of a generic LDS pointer are the LDS byte offset per the aperture map).
// The async-to-LDS builtin takes v4i32 pointers in AS1 (src) and AS3 (dst).
#define TO_LDSOFF(p) ((unsigned)(unsigned long long)(p))
#define TO_GLBV4(p) ((__attribute__((address_space(1))) i32x4*)(unsigned long long)(p))
#define TO_LDSV4(p) ((__attribute__((address_space(3))) i32x4*)(unsigned)(unsigned long long)(p))

__device__ __forceinline__ unsigned short f32_to_bf16_bits(float f) {
  unsigned u = __builtin_bit_cast(unsigned, f);
  u += 0x7FFFu + ((u >> 16) & 1u);          // round-to-nearest-even
  return (unsigned short)(u >> 16);
}
__device__ __forceinline__ __bf16 f32_to_bf16(float f) {
  unsigned short s = f32_to_bf16_bits(f);
  return __builtin_bit_cast(__bf16, s);
}
__device__ __forceinline__ float bf16_to_f32(__bf16 h) {
  unsigned u = ((unsigned)__builtin_bit_cast(unsigned short, h)) << 16;
  return __builtin_bit_cast(float, u);
}
__device__ __forceinline__ unsigned pack_bf16x2(float a, float b) {
  return (unsigned)f32_to_bf16_bits(a) | ((unsigned)f32_to_bf16_bits(b) << 16);
}

// 16-bit A/B fragment (16x32, row-major in LDS with leading dim 32):
// lanes 0-15: row = lane,    VGPR0-3 -> K 0..7,  VGPR4-7 -> K 16..23
// lanes 16-31: row = lane-16, VGPR0-3 -> K 8..15, VGPR4-7 -> K 24..31
__device__ __forceinline__ v16bf load_frag32(const __bf16* base, int lane) {
  const int r  = lane & 15;
  const int kb = (lane >> 4) << 3;   // 0 or 8
  v8bf lo = *(const v8bf*)(base + r * 32 + kb);
  v8bf hi = *(const v8bf*)(base + r * 32 + 16 + kb);
  v16bf out;
#pragma unroll
  for (int i = 0; i < 8; ++i) { out[i] = lo[i]; out[i + 8] = hi[i]; }
  return out;
}

#ifdef HAVE_TDM
// 2D bf16 tile DMA: tile_rows x tile_cols (contiguous), row stride in elements.
// D# group0: count=1 | lds_addr | global_addr[56:0] | type=2
// D# group1: data_size=2B | tensor_dim0/1 | tile_dim0/1 | tensor_dim0_stride
__device__ __forceinline__ void tdm_load_tile_bf16(
    unsigned lds_byte_off, const __bf16* gsrc,
    unsigned tile_cols, unsigned tile_rows, unsigned row_stride_elems)
{
  unsigned long long ga = (unsigned long long)gsrc;
  u32x4 g0;
  g0[0] = 1u;                                                   // 1 descriptor, user mode
  g0[1] = lds_byte_off;                                         // LDS byte address
  g0[2] = (unsigned)ga;                                         // global_addr[31:0]
  g0[3] = (unsigned)((ga >> 32) & 0x01FFFFFFull) | (2u << 30);  // addr[56:32] | type=2
  i32x8 g1;
  g1[0] = (int)(1u << 16);                                      // data_size code 1 (2 bytes)
  g1[1] = (int)((row_stride_elems & 0xFFFFu) << 16);            // tensor_dim0[15:0]
  g1[2] = (int)((row_stride_elems >> 16) | (tile_rows << 16));  // dim0 hi | tensor_dim1 lo
  g1[3] = (int)((tile_cols & 0xFFFFu) << 16);                   // dim1 hi(0) | tile_dim0
  g1[4] = (int)(tile_rows & 0xFFFFu);                           // tile_dim1 | tile_dim2=0
  g1[5] = (int)row_stride_elems;                                // tensor_dim0_stride[31:0]
  g1[6] = 0;
  g1[7] = 0;
  i32x4 z4 = {0, 0, 0, 0};
#ifdef CDNA5_TDM_6ARG
  i32x8 z8 = {0, 0, 0, 0, 0, 0, 0, 0};
  __builtin_amdgcn_tensor_load_to_lds(g0, g1, z4, z4, z8, 0);
#else
  __builtin_amdgcn_tensor_load_to_lds(g0, g1, z4, z4, 0);
#endif
}
#endif

// ================= GEMM 1: q = relu([uX|cS] @ W^T + b), bf16 out =================
// tile: 128(M) x 64(N), K step 32; 8 waves in 4x2, each wave 32x32 (2x2 WMMA)
__global__ __launch_bounds__(256) void k_gemm_q(
    const float* __restrict__ uX, const float* __restrict__ cS,
    const float* __restrict__ W,  const float* __restrict__ bias,
    __bf16* __restrict__ qbf)
{
  __shared__ __bf16 As[128][32];
  __shared__ __bf16 Bs[64][32];

  const int tid  = threadIdx.x;
  const int lane = tid & 31;
  const int wave = tid >> 5;
  const int wm   = wave >> 1;        // 0..3
  const int wn   = wave & 1;         // 0..1
  const int m0   = blockIdx.y * 128;
  const int n0   = blockIdx.x * 64;

  v8f acc[2][2] = {};

  const int ar = tid >> 1;           // 0..127
  const int ac = (tid & 1) * 16;
  const int br = tid >> 2;           // 0..63
  const int bc = (tid & 3) * 8;

  for (int k0 = 0; k0 < OBS + PLAN; k0 += 32) {
    const float* Xsrc = (k0 < OBS) ? (uX + (size_t)m0 * OBS + k0)
                                   : (cS + (size_t)m0 * PLAN + (k0 - OBS));
    // stage A: 128x32, f32 -> bf16 packed, 16 elems/thread
    {
      const float4* src = (const float4*)(Xsrc + (size_t)ar * OBS + ac);
      __builtin_prefetch((const void*)(src + 8), 0, 0);   // next K tile of this row
      unsigned pk[8];
#pragma unroll
      for (int i = 0; i < 4; ++i) {
        float4 f = src[i];
        pk[2 * i + 0] = pack_bf16x2(f.x, f.y);
        pk[2 * i + 1] = pack_bf16x2(f.z, f.w);
      }
      uint4* dst = (uint4*)&As[ar][ac];
      dst[0] = make_uint4(pk[0], pk[1], pk[2], pk[3]);
      dst[1] = make_uint4(pk[4], pk[5], pk[6], pk[7]);
    }
    // stage B: W is [QD][OBS+PLAN] row-major; B tile stored [n][k] (already matches)
    {
      const float4* src = (const float4*)(W + (size_t)(n0 + br) * (OBS + PLAN) + k0 + bc);
      __builtin_prefetch((const void*)(src + 8), 0, 0);
      float4 f0 = src[0], f1 = src[1];
      *(uint4*)&Bs[br][bc] = make_uint4(pack_bf16x2(f0.x, f0.y), pack_bf16x2(f0.z, f0.w),
                                        pack_bf16x2(f1.x, f1.y), pack_bf16x2(f1.z, f1.w));
    }
    __syncthreads();

    v16bf afrag[2], bfrag[2];
    afrag[0] = load_frag32(&As[wm * 32 + 0][0], lane);
    afrag[1] = load_frag32(&As[wm * 32 + 16][0], lane);
    bfrag[0] = load_frag32(&Bs[wn * 32 + 0][0], lane);
    bfrag[1] = load_frag32(&Bs[wn * 32 + 16][0], lane);
#pragma unroll
    for (int i = 0; i < 2; ++i)
#pragma unroll
      for (int j = 0; j < 2; ++j)
        acc[i][j] = __builtin_amdgcn_wmma_f32_16x16x32_bf16(
            false, afrag[i], false, bfrag[j], (short)0, acc[i][j], false, false);
    __syncthreads();
  }

  // epilogue: bias + relu, store bf16
  const int ncl   = lane & 15;
  const int rbase = (lane < 16) ? 0 : 8;
#pragma unroll
  for (int i = 0; i < 2; ++i)
#pragma unroll
    for (int j = 0; j < 2; ++j) {
      const int col = n0 + wn * 32 + j * 16 + ncl;
      const float bv = bias[col];
#pragma unroll
      for (int v = 0; v < 8; ++v) {
        const int row = m0 + wm * 32 + i * 16 + rbase + v;
        float val = fmaxf(acc[i][j][v] + bv, 0.0f);
        qbf[(size_t)row * QD + col] = f32_to_bf16(val);
      }
    }
}

// ================= GEMM 2: scores = (q . k) * qinv * kinv / TEMP =================
// double-buffered LDS; staged by TDM (preferred), async-to-LDS, or sync copies
#ifndef HAVE_TDM
__device__ __forceinline__ void stage_scores_tile(
    __bf16 (*As)[32], __bf16 (*Bs)[32],
    const __bf16* qsrc, const __bf16* ksrc, int tid)
{
  const int ar = tid >> 1, ac = (tid & 1) * 16;
  const int br = tid >> 2, bc = (tid & 3) * 8;
#ifdef HAVE_ASYNC_LDS
  __builtin_amdgcn_global_load_async_to_lds_b128(
      TO_GLBV4(qsrc + (size_t)ar * QD + ac),     TO_LDSV4(&As[ar][ac]),     0, 0);
  __builtin_amdgcn_global_load_async_to_lds_b128(
      TO_GLBV4(qsrc + (size_t)ar * QD + ac + 8), TO_LDSV4(&As[ar][ac + 8]), 0, 0);
  __builtin_amdgcn_global_load_async_to_lds_b128(
      TO_GLBV4(ksrc + (size_t)br * QD + bc),     TO_LDSV4(&Bs[br][bc]),     0, 0);
#else
  const uint4* sa = (const uint4*)(qsrc + (size_t)ar * QD + ac);
  uint4* da = (uint4*)&As[ar][ac];
  da[0] = sa[0];
  da[1] = sa[1];
  *(uint4*)&Bs[br][bc] = *(const uint4*)(ksrc + (size_t)br * QD + bc);
#endif
}
#endif

__global__ __launch_bounds__(256) void k_gemm_scores(
    const __bf16* __restrict__ qbf, const __bf16* __restrict__ kbf,
    const float* __restrict__ qinv, const float* __restrict__ kinv,
    float* __restrict__ scores)
{
  __shared__ __bf16 As[2][128][32];
  __shared__ __bf16 Bs[2][64][32];

  const int tid  = threadIdx.x;
  const int lane = tid & 31;
  const int wave = tid >> 5;
  const int wm   = wave >> 1;
  const int wn   = wave & 1;
  const int m0   = blockIdx.y * 128;
  const int n0   = blockIdx.x * 64;

  v8f acc[2][2] = {};
  const int NSTEP = QD / 32;

  // ---- prologue: fill buffer 0 ----
#ifdef HAVE_TDM
  // each wave DMAs its own slice: 16 A rows + 8 B rows (wave-uniform descriptors)
  tdm_load_tile_bf16(TO_LDSOFF(&As[0][wave * 16][0]),
                     qbf + (size_t)(m0 + wave * 16) * QD, 32, 16, QD);
  tdm_load_tile_bf16(TO_LDSOFF(&Bs[0][wave * 8][0]),
                     kbf + (size_t)(n0 + wave * 8) * QD, 32, 8, QD);
#else
  stage_scores_tile(As[0], Bs[0], qbf + (size_t)m0 * QD, kbf + (size_t)n0 * QD, tid);
#endif

  for (int s = 0; s < NSTEP; ++s) {
    const int buf = s & 1;
    const int k0 = s * 32;
#ifdef HAVE_TDM
    if (s + 1 < NSTEP) {
      tdm_load_tile_bf16(TO_LDSOFF(&As[buf ^ 1][wave * 16][0]),
                         qbf + (size_t)(m0 + wave * 16) * QD + k0 + 32, 32, 16, QD);
      tdm_load_tile_bf16(TO_LDSOFF(&Bs[buf ^ 1][wave * 8][0]),
                         kbf + (size_t)(n0 + wave * 8) * QD + k0 + 32, 32, 8, QD);
      WAIT_TENSOR(2);     // previous buffer's pair complete; next pair in flight
    } else {
      WAIT_TENSOR(0);
    }
    __syncthreads();
#else
    if (s + 1 < NSTEP) {
      stage_scores_tile(As[buf ^ 1], Bs[buf ^ 1],
                        qbf + (size_t)m0 * QD + k0 + 32,
                        kbf + (size_t)n0 * QD + k0 + 32, tid);
#ifdef HAVE_ASYNC_LDS
      WAIT_ASYNC(3);      // wait for previous tile's 3 async loads
#endif
    } else {
#ifdef HAVE_ASYNC_LDS
      WAIT_ASYNC(0);
#endif
    }
    __syncthreads();
#endif

    v16bf afrag[2], bfrag[2];
    afrag[0] = load_frag32(&As[buf][wm * 32 + 0][0], lane);
    afrag[1] = load_frag32(&As[buf][wm * 32 + 16][0], lane);
    bfrag[0] = load_frag32(&Bs[buf][wn * 32 + 0][0], lane);
    bfrag[1] = load_frag32(&Bs[buf][wn * 32 + 16][0], lane);
#pragma unroll
    for (int i = 0; i < 2; ++i)
#pragma unroll
      for (int j = 0; j < 2; ++j)
        acc[i][j] = __builtin_amdgcn_wmma_f32_16x16x32_bf16(
            false, afrag[i], false, bfrag[j], (short)0, acc[i][j], false, false);
    __syncthreads();
  }

  const int ncl   = lane & 15;
  const int rbase = (lane < 16) ? 0 : 8;
#pragma unroll
  for (int i = 0; i < 2; ++i)
#pragma unroll
    for (int j = 0; j < 2; ++j) {
      const int col = n0 + wn * 32 + j * 16 + ncl;
      const float ki = kinv[col] * INV_TEMP;
      const bool valid = (col < KB);
#pragma unroll
      for (int v = 0; v < 8; ++v) {
        const int row = m0 + wm * 32 + i * 16 + rbase + v;
        float sc = acc[i][j][v] * qinv[row] * ki;
        scores[(size_t)row * KBP + col] = valid ? sc : NEGBIG;
      }
    }
}

// ================= norms / key prep / init =================
__global__ __launch_bounds__(256) void k_rownorm_f32(
    const float* __restrict__ x, float* __restrict__ inv, int n)
{
  __shared__ float red[256];
  const int row = blockIdx.x;
  const float* p = x + (size_t)row * n;
  float ss = 0.f;
  for (int c = threadIdx.x; c < n; c += 256) { float v = p[c]; ss += v * v; }
  red[threadIdx.x] = ss; __syncthreads();
  for (int s = 128; s > 0; s >>= 1) {
    if (threadIdx.x < s) red[threadIdx.x] += red[threadIdx.x + s];
    __syncthreads();
  }
  if (threadIdx.x == 0) inv[row] = 1.0f / fmaxf(sqrtf(red[0]), EPSN);
}

__global__ __launch_bounds__(256) void k_rownorm_bf16(
    const __bf16* __restrict__ x, float* __restrict__ inv, int n)
{
  __shared__ float red[256];
  const int row = blockIdx.x;
  const __bf16* p = x + (size_t)row * n;
  float ss = 0.f;
  for (int c = threadIdx.x; c < n; c += 256) { float v = bf16_to_f32(p[c]); ss += v * v; }
  red[threadIdx.x] = ss; __syncthreads();
  for (int s = 128; s > 0; s >>= 1) {
    if (threadIdx.x < s) red[threadIdx.x] += red[threadIdx.x + s];
    __syncthreads();
  }
  if (threadIdx.x == 0) inv[row] = 1.0f / fmaxf(sqrtf(red[0]), EPSN);
}

__global__ __launch_bounds__(256) void k_keyprep(
    const float* __restrict__ keys, __bf16* __restrict__ kbf, float* __restrict__ kinv)
{
  __shared__ float red[256];
  const int r = blockIdx.x;
  const int c0 = threadIdx.x * 8;          // 256*8 == 2048 == QD
  if (r >= KB) {
    *(uint4*)&kbf[(size_t)r * QD + c0] = make_uint4(0u, 0u, 0u, 0u);
    if (threadIdx.x == 0) kinv[r] = 0.f;
    return;
  }
  const float4* src = (const float4*)(keys + (size_t)r * QD + c0);
  float4 f0 = src[0], f1 = src[1];
  *(uint4*)&kbf[(size_t)r * QD + c0] =
      make_uint4(pack_bf16x2(f0.x, f0.y), pack_bf16x2(f0.z, f0.w),
                 pack_bf16x2(f1.x, f1.y), pack_bf16x2(f1.z, f1.w));
  float ss = f0.x * f0.x + f0.y * f0.y + f0.z * f0.z + f0.w * f0.w
           + f1.x * f1.x + f1.y * f1.y + f1.z * f1.z + f1.w * f1.w;
  red[threadIdx.x] = ss; __syncthreads();
  for (int s = 128; s > 0; s >>= 1) {
    if (threadIdx.x < s) red[threadIdx.x] += red[threadIdx.x + s];
    __syncthreads();
  }
  if (threadIdx.x == 0) kinv[r] = 1.0f / fmaxf(sqrtf(red[0]), EPSN);
}

__global__ void k_init(float* __restrict__ cost) { if (threadIdx.x == 0) *cost = 0.f; }

// ================= top-k + softmax + gather + alignment =================
__global__ __launch_bounds__(256) void k_topk(
    const float* __restrict__ scores, const float* __restrict__ values,
    const float* __restrict__ cS, const float* __restrict__ sem,
    const float* __restrict__ csinv, const float* __restrict__ seminv,
    float* __restrict__ outE, float* __restrict__ outIdx, float* __restrict__ outCost)
{
  __shared__ float s[KBP];
  __shared__ float rv[256];
  __shared__ int   ri[256];
  __shared__ float a_sh[TOPK];
  __shared__ int   i_sh[TOPK];
  __shared__ float dsh[TOPK];

  const int b = blockIdx.x;
  const int tid = threadIdx.x;
  const float* srow = scores + (size_t)b * KBP;

#ifdef HAVE_ASYNC_LDS
  // 256 threads x 16B = 4KB row staged asynchronously to LDS
  __builtin_amdgcn_global_load_async_to_lds_b128(
      TO_GLBV4(srow + 4 * tid), TO_LDSV4(&s[4 * tid]), 0, 0);
  WAIT_ASYNC(0);
#else
  for (int c = tid; c < KBP; c += 256) s[c] = srow[c];
#endif
  __syncthreads();

  // 5 rounds of argmax with masking
  for (int t = 0; t < TOPK; ++t) {
    float bv = NEGBIG; int bi = 0;
    for (int c = tid; c < KBP; c += 256) {
      float v = s[c];
      if (v > bv) { bv = v; bi = c; }
    }
    rv[tid] = bv; ri[tid] = bi; __syncthreads();
    for (int st = 128; st > 0; st >>= 1) {
      if (tid < st && rv[tid + st] > rv[tid]) { rv[tid] = rv[tid + st]; ri[tid] = ri[tid + st]; }
      __syncthreads();
    }
    if (tid == 0) { a_sh[t] = rv[0]; i_sh[t] = ri[0]; s[ri[0]] = NEGBIG; }
    __syncthreads();
  }

  // softmax over top-5
  if (tid == 0) {
    float mx = a_sh[0];
    float e[TOPK], sum = 0.f;
#pragma unroll
    for (int t = 0; t < TOPK; ++t) { e[t] = __expf(a_sh[t] - mx); sum += e[t]; }
    float is = 1.0f / sum;
#pragma unroll
    for (int t = 0; t < TOPK; ++t) a_sh[t] = e[t] * is;
  }
  __syncthreads();

  float al[TOPK]; int ix[TOPK];
#pragma unroll
  for (int t = 0; t < TOPK; ++t) { al[t] = a_sh[t]; ix[t] = i_sh[t]; }

  // E = sum_t alpha_t * values[idx_t]
  for (int c = tid; c < VD; c += 256) {
    float e = 0.f;
#pragma unroll
    for (int t = 0; t < TOPK; ++t) e += al[t] * values[(size_t)ix[t] * VD + c];
    outE[(size_t)b * VD + c] = e;
  }

  // alignment dots: dot(c_S[b], sem[idx_t])
  float pd[TOPK] = {0.f, 0.f, 0.f, 0.f, 0.f};
  for (int c = tid; c < PLAN; c += 256) {
    float cv = cS[(size_t)b * PLAN + c];
#pragma unroll
    for (int t = 0; t < TOPK; ++t) pd[t] += cv * sem[(size_t)ix[t] * PLAN + c];
  }
#pragma unroll
  for (int t = 0; t < TOPK; ++t) {
    rv[tid] = pd[t]; __syncthreads();
    for (int st = 128; st > 0; st >>= 1) {
      if (tid < st) rv[tid] += rv[tid + st];
      __syncthreads();
    }
    if (tid == 0) dsh[t] = rv[0];
    __syncthreads();
  }

  if (tid == 0) {
    float ci = csinv[b], cost = 0.f;
#pragma unroll
    for (int t = 0; t < TOPK; ++t) {
      float cosv = dsh[t] * ci * seminv[ix[t]];
      cost += al[t] * (1.0f - cosv);
    }
    atomicAdd(outCost, cost * (1.0f / (float)BATCH));
  }
  if (tid < TOPK) outIdx[(size_t)b * TOPK + tid] = (float)i_sh[tid];
}

// ================= host launcher =================
extern "C" void kernel_launch(void* const* d_in, const int* in_sizes, int n_in,
                              void* d_out, int out_size, void* d_ws, size_t ws_size,
                              hipStream_t stream) {
  (void)in_sizes; (void)n_in; (void)out_size; (void)ws_size;
  const float* uX   = (const float*)d_in[0];
  const float* cS   = (const float*)d_in[1];
  const float* W    = (const float*)d_in[2];
  const float* bias = (const float*)d_in[3];
  const float* keys = (const float*)d_in[4];
  const float* vals = (const float*)d_in[5];
  const float* sem  = (const float*)d_in[6];

  float* out      = (float*)d_out;
  float* outE     = out;
  float* outIdx   = out + (size_t)BATCH * VD;
  float* outCost  = out + (size_t)BATCH * VD + (size_t)BATCH * TOPK;

  char* w = (char*)d_ws;
  __bf16* qbf    = (__bf16*)w;  w += (size_t)BATCH * QD * 2;
  __bf16* kbf    = (__bf16*)w;  w += (size_t)KBP * QD * 2;
  float*  scores = (float*)w;   w += (size_t)BATCH * KBP * 4;
  float*  qinv   = (float*)w;   w += (size_t)BATCH * 4;
  float*  kinv   = (float*)w;   w += (size_t)KBP * 4;
  float*  csinv  = (float*)w;   w += (size_t)BATCH * 4;
  float*  seminv = (float*)w;   w += (size_t)KBP * 4;

  hipLaunchKernelGGL(k_init, dim3(1), dim3(32), 0, stream, outCost);
  hipLaunchKernelGGL(k_keyprep, dim3(KBP), dim3(256), 0, stream, keys, kbf, kinv);
  hipLaunchKernelGGL(k_rownorm_f32, dim3(BATCH), dim3(256), 0, stream, cS, csinv, PLAN);
  hipLaunchKernelGGL(k_rownorm_f32, dim3(KB), dim3(256), 0, stream, sem, seminv, PLAN);
  hipLaunchKernelGGL(k_gemm_q, dim3(QD / 64, BATCH / 128), dim3(256), 0, stream,
                     uX, cS, W, bias, qbf);
  hipLaunchKernelGGL(k_rownorm_bf16, dim3(BATCH), dim3(256), 0, stream, qbf, qinv, QD);
  hipLaunchKernelGGL(k_gemm_scores, dim3(KBP / 64, BATCH / 128), dim3(256), 0, stream,
                     qbf, kbf, qinv, kinv, scores);
  hipLaunchKernelGGL(k_topk, dim3(BATCH), dim3(256), 0, stream,
                     scores, vals, cS, sem, csinv, seminv, outE, outIdx, outCost);
}
